// VSSBlock3D_v6_50371376447798
// MI455X (gfx1250) — compile-verified, matrix operand order checked by reading, not run on hardware
//
#include <hip/hip_runtime.h>
#include <hip/hip_bf16.h>

typedef __bf16 bf16;
typedef __attribute__((ext_vector_type(16))) __bf16 v16bf;
typedef __attribute__((ext_vector_type(8)))  __bf16 v8bf;
typedef __attribute__((ext_vector_type(4)))  __bf16 v4bf;
typedef __attribute__((ext_vector_type(8)))  float  v8f;

#define BM 64
#define BN 128
#define BK 32

// ISA 7.12.2 16-bit fragment mapping for a 16x32 subtile, element (m,k):
//   lane = m + 16*((k>>3)&1),  e = (k>>4)*8 + (k&7)
// We store LDS tiles pre-swizzled as frag[sub][lane][e] so each lane's whole
// fragment is 32 contiguous bytes (-> 2x ds_load_b128).

__device__ __forceinline__ float apply_act(float v, int act) {
    if (act == 1) {                       // softplus
        v = (v > 20.0f) ? v : log1pf(__expf(v));
    } else if (act == 2) {                // gelu (tanh approx)
        float c = v + 0.044715f * v * v * v;
        v = 0.5f * v * (1.0f + tanhf(0.7978845608f * c));
    }
    return v;
}

// D = act(A@B + bias) + residual.  A: MxK f32 (lda), B: KxN f32 (ldb, row major),
// D/residual: MxN f32 (ldd).  Block tile 64x128, wave tile 32x32, bf16 WMMA.
template <bool FULL>
__global__ __launch_bounds__(256) void gemm_wmma_kernel(
    const float* __restrict__ A, int lda,
    const float* __restrict__ B, int ldb,
    const float* __restrict__ bias,
    const float* __restrict__ residual,
    float* __restrict__ D, int ldd,
    int M, int N, int K, int act)
{
    __shared__ __align__(32) bf16 Afrag[2][4][32][16];   // [buf][msub][lane][e]
    __shared__ __align__(32) bf16 Bfrag[2][8][32][16];   // [buf][nsub][lane][e]

    const int tid  = threadIdx.x;
    const int lane = tid & 31;
    const int wid  = tid >> 5;
    const int m0   = blockIdx.y * BM;
    const int n0   = blockIdx.x * BN;
    const int wms  = (wid & 1) * 2;        // wave's first m-subtile (of 4)
    const int wns  = (wid >> 1) * 2;       // wave's first n-subtile (of 8)

    const int khalf = lane >> 4;
    const int l16   = lane & 15;

    // ---- cooperative-load thread mappings ----
    // A: 2 chunks of float4; chunk q covers idx = tid*4 + q*1024 -> (m, k0)
    const int a_m[2]  = { (tid * 4) >> 5, (tid * 4 + 1024) >> 5 };
    const int a_k0    = (tid * 4) & 31;          // same for both chunks
    // B: one column n, 16 consecutive k starting at kq
    const int b_n  = tid & 127;
    const int b_kq = (tid >> 7) * 16;

    float ar[2][4];
    float br[16];

    auto gload = [&](int kk) {
        if (FULL) {
            #pragma unroll
            for (int q = 0; q < 2; ++q) {
                const float4 v = *(const float4*)(A + (size_t)(m0 + a_m[q]) * lda + kk + a_k0);
                ar[q][0] = v.x; ar[q][1] = v.y; ar[q][2] = v.z; ar[q][3] = v.w;
            }
            #pragma unroll
            for (int j = 0; j < 16; ++j)
                br[j] = B[(size_t)(kk + b_kq + j) * ldb + n0 + b_n];
        } else {
            #pragma unroll
            for (int q = 0; q < 2; ++q)
                #pragma unroll
                for (int j = 0; j < 4; ++j) {
                    int gm = m0 + a_m[q], gk = kk + a_k0 + j;
                    int cm = gm < M ? gm : M - 1;
                    int ck = gk < K ? gk : K - 1;
                    float v = A[(size_t)cm * lda + ck];
                    ar[q][j] = (gm < M && gk < K) ? v : 0.0f;
                }
            int gn = n0 + b_n;
            int cn = gn < N ? gn : N - 1;
            #pragma unroll
            for (int j = 0; j < 16; ++j) {
                int gk = kk + b_kq + j;
                int ck = gk < K ? gk : K - 1;
                float v = B[(size_t)ck * ldb + cn];
                br[j] = (gk < K && gn < N) ? v : 0.0f;
            }
        }
    };

    auto lstore = [&](int buf) {
        // A: each chunk = 4 consecutive k (aligned 4) -> 4 consecutive e's
        #pragma unroll
        for (int q = 0; q < 2; ++q) {
            int msub = a_m[q] >> 4, ml = a_m[q] & 15;
            int la   = ml + ((a_k0 >> 3) & 1) * 16;
            int e0   = (a_k0 >> 4) * 8 + (a_k0 & 7);
            v4bf t;
            t[0] = (bf16)ar[q][0]; t[1] = (bf16)ar[q][1];
            t[2] = (bf16)ar[q][2]; t[3] = (bf16)ar[q][3];
            *(v4bf*)&Afrag[buf][msub][la][e0] = t;
        }
        // B: k = kq..kq+7 -> lane nl (khalf 0), k = kq+8..kq+15 -> lane nl+16
        {
            int nsub = b_n >> 4, nl = b_n & 15;
            int eb = (b_kq >> 4) * 8;
            v8bf lo, hi;
            #pragma unroll
            for (int j = 0; j < 8; ++j) { lo[j] = (bf16)br[j]; hi[j] = (bf16)br[8 + j]; }
            *(v8bf*)&Bfrag[buf][nsub][nl][eb]      = lo;
            *(v8bf*)&Bfrag[buf][nsub][nl + 16][eb] = hi;
        }
    };

    v8f acc00 = {}, acc01 = {}, acc10 = {}, acc11 = {};

    // prologue: tile 0 -> LDS[0]
    gload(0);
    lstore(0);
    __syncthreads();

    int buf = 0;
    for (int kk = 0; kk < K; kk += BK) {
        const bool more = (kk + BK) < K;
        if (more) gload(kk + BK);            // issue next tile's global loads

        const v16bf a0 = *(const v16bf*)&Afrag[buf][wms + 0][lane][0];
        const v16bf a1 = *(const v16bf*)&Afrag[buf][wms + 1][lane][0];
        const v16bf b0 = *(const v16bf*)&Bfrag[buf][wns + 0][lane][0];
        const v16bf b1 = *(const v16bf*)&Bfrag[buf][wns + 1][lane][0];

        acc00 = __builtin_amdgcn_wmma_f32_16x16x32_bf16(false, a0, false, b0, (short)0, acc00, false, false);
        acc01 = __builtin_amdgcn_wmma_f32_16x16x32_bf16(false, a0, false, b1, (short)0, acc01, false, false);
        acc10 = __builtin_amdgcn_wmma_f32_16x16x32_bf16(false, a1, false, b0, (short)0, acc10, false, false);
        acc11 = __builtin_amdgcn_wmma_f32_16x16x32_bf16(false, a1, false, b1, (short)0, acc11, false, false);

        if (more) lstore(buf ^ 1);           // stage next tile behind the MMAs
        __syncthreads();
        buf ^= 1;
    }

    // ---- epilogue: D layout -> VGPR r: M = r (+8 upper lane half), lane = N
    const int wm0 = m0 + (wid & 1) * 32;
    const int wn0 = n0 + (wid >> 1) * 32;
    #pragma unroll
    for (int mt = 0; mt < 2; ++mt) {
        #pragma unroll
        for (int r = 0; r < 8; ++r) {
            int m = wm0 + mt * 16 + r + khalf * 8;
            if (!FULL && m >= M) continue;
            #pragma unroll
            for (int nt = 0; nt < 2; ++nt) {
                int n = wn0 + nt * 16 + l16;
                if (!FULL && n >= N) continue;
                float v = (mt == 0) ? (nt == 0 ? acc00[r] : acc01[r])
                                    : (nt == 0 ? acc10[r] : acc11[r]);
                if (bias) v += bias[n];
                v = apply_act(v, act);
                if (residual) v += residual[(size_t)m * ldd + n];
                D[(size_t)m * ldd + n] = v;
            }
        }
    }
}

// Row-wise LayerNorm over C (block = 128 threads, one row/block)
__global__ __launch_bounds__(128) void ln_kernel(
    const float* __restrict__ x, const float* __restrict__ g,
    const float* __restrict__ b, float* __restrict__ out, int C)
{
    const int row = blockIdx.x;
    const int tid = threadIdx.x;
    __shared__ float red[128];
    const float* xr = x + (size_t)row * C;

    float s = 0.f;
    for (int c = tid; c < C; c += 128) s += xr[c];
    red[tid] = s; __syncthreads();
    for (int o = 64; o > 0; o >>= 1) { if (tid < o) red[tid] += red[tid + o]; __syncthreads(); }
    float mean = red[0] / (float)C;
    __syncthreads();

    float vs = 0.f;
    for (int c = tid; c < C; c += 128) { float d = xr[c] - mean; vs += d * d; }
    red[tid] = vs; __syncthreads();
    for (int o = 64; o > 0; o >>= 1) { if (tid < o) red[tid] += red[tid + o]; __syncthreads(); }
    float rstd = rsqrtf(red[0] / (float)C + 1e-6f);

    for (int c = tid; c < C; c += 128)
        out[(size_t)row * C + c] = (xr[c] - mean) * rstd * g[c] + b[c];
}

// Depthwise 3x3x3 conv (SAME, zero pad) + bias + SiLU over [B=2,8,16,16] grid,
// reading xin = xz[:, 0:384] of the [4096,768] buffer, writing [4096,384].
__global__ __launch_bounds__(256) void conv_dw_silu_kernel(
    const float* __restrict__ xz, const float* __restrict__ w,
    const float* __restrict__ bias, float* __restrict__ out)
{
    int idx = blockIdx.x * 256 + threadIdx.x;
    if (idx >= 4096 * 384) return;
    int c   = idx % 384;
    int row = idx / 384;
    int bb  = row >> 11;
    int rl  = row & 2047;
    int dz  = rl >> 8;
    int hy  = (rl >> 4) & 15;
    int wx  = rl & 15;

    float acc = bias[c];
    #pragma unroll
    for (int kd = 0; kd < 3; ++kd) {
        int z = dz + kd - 1; if (z < 0 || z >= 8) continue;
        #pragma unroll
        for (int kh = 0; kh < 3; ++kh) {
            int y = hy + kh - 1; if (y < 0 || y >= 16) continue;
            #pragma unroll
            for (int kw = 0; kw < 3; ++kw) {
                int xq = wx + kw - 1; if (xq < 0 || xq >= 16) continue;
                int r2 = bb * 2048 + z * 256 + y * 16 + xq;
                acc += xz[(size_t)r2 * 768 + c] * w[((c * 3 + kd) * 3 + kh) * 3 + kw];
            }
        }
    }
    out[(size_t)row * 384 + c] = acc / (1.0f + __expf(-acc));   // SiLU
}

// Selective scan: lanes s=0..15 per half-wave own state h[d][s]; y via shfl_xor
// reduction (masks 1/2/4/8 stay inside each 16-lane group on wave32).
__global__ __launch_bounds__(256) void scan_kernel(
    const float* __restrict__ dt,    // [4096,384] softplus'ed
    const float* __restrict__ xc,    // [4096,384] conv+silu branch
    const float* __restrict__ dbc,   // [4096,64]: Bm cols 24..39, Cm cols 40..55
    const float* __restrict__ A_log, // [384,16]
    float* __restrict__ ys)          // [4096,384]
{
    const int tid  = threadIdx.x;
    const int lane = tid & 31;
    const int wave = tid >> 5;
    const int s    = lane & 15;
    const int half = lane >> 4;
    const int d    = blockIdx.x * 16 + wave * 2 + half;   // channel
    const int bb   = blockIdx.y;                          // batch

    const float a = -__expf(A_log[d * 16 + s]);
    float h = 0.0f;
    for (int l = 0; l < 2048; ++l) {
        const size_t row = (size_t)bb * 2048 + l;
        float dtv = dt[row * 384 + d];
        float xv  = xc[row * 384 + d];
        float bm  = dbc[row * 64 + 24 + s];
        float cm  = dbc[row * 64 + 40 + s];
        h = __expf(dtv * a) * h + dtv * bm * xv;
        float p = h * cm;
        p += __shfl_xor(p, 1);
        p += __shfl_xor(p, 2);
        p += __shfl_xor(p, 4);
        p += __shfl_xor(p, 8);
        if (s == 0) ys[row * 384 + d] = p;
    }
}

// y = LN(ys + xc*D_skip; on_g,on_b) * silu(z) ; z = xz[:,384:768]
__global__ __launch_bounds__(128) void gate_kernel(
    const float* __restrict__ ys, const float* __restrict__ xc,
    const float* __restrict__ Dsk, const float* __restrict__ xz,
    const float* __restrict__ g, const float* __restrict__ b,
    float* __restrict__ out)
{
    const int row = blockIdx.x;
    const int tid = threadIdx.x;
    __shared__ float red[128];

    float yv[3];
    #pragma unroll
    for (int i = 0; i < 3; ++i) {
        int c = tid + i * 128;
        yv[i] = ys[(size_t)row * 384 + c] + xc[(size_t)row * 384 + c] * Dsk[c];
    }
    red[tid] = yv[0] + yv[1] + yv[2]; __syncthreads();
    for (int o = 64; o > 0; o >>= 1) { if (tid < o) red[tid] += red[tid + o]; __syncthreads(); }
    float mean = red[0] * (1.0f / 384.0f);
    __syncthreads();

    float vs = 0.f;
    #pragma unroll
    for (int i = 0; i < 3; ++i) { float dd = yv[i] - mean; vs += dd * dd; }
    red[tid] = vs; __syncthreads();
    for (int o = 64; o > 0; o >>= 1) { if (tid < o) red[tid] += red[tid + o]; __syncthreads(); }
    float rstd = rsqrtf(red[0] * (1.0f / 384.0f) + 1e-6f);

    #pragma unroll
    for (int i = 0; i < 3; ++i) {
        int c = tid + i * 128;
        float z  = xz[(size_t)row * 768 + 384 + c];
        float sz = z / (1.0f + __expf(-z));
        out[(size_t)row * 384 + c] = ((yv[i] - mean) * rstd * g[c] + b[c]) * sz;
    }
}

extern "C" void kernel_launch(void* const* d_in, const int* in_sizes, int n_in,
                              void* d_out, int out_size, void* d_ws, size_t ws_size,
                              hipStream_t stream) {
    (void)in_sizes; (void)n_in; (void)out_size; (void)ws_size;
    const float* x     = (const float*)d_in[0];
    const float* g1    = (const float*)d_in[1];
    const float* b1    = (const float*)d_in[2];
    const float* W_in  = (const float*)d_in[3];
    const float* b_in  = (const float*)d_in[4];
    const float* convw = (const float*)d_in[5];
    const float* convb = (const float*)d_in[6];
    const float* W_x   = (const float*)d_in[7];
    const float* W_dt  = (const float*)d_in[8];
    const float* b_dt  = (const float*)d_in[9];
    const float* A_log = (const float*)d_in[10];
    const float* Dsk   = (const float*)d_in[11];
    const float* on_g  = (const float*)d_in[12];
    const float* on_b  = (const float*)d_in[13];
    const float* W_out = (const float*)d_in[14];
    const float* b_out = (const float*)d_in[15];
    const float* g2    = (const float*)d_in[16];
    const float* b2    = (const float*)d_in[17];
    const float* W1    = (const float*)d_in[18];
    const float* b1m   = (const float*)d_in[19];
    const float* W2    = (const float*)d_in[20];
    const float* b2m   = (const float*)d_in[21];
    float* out = (float*)d_out;

    const int R = 4096, C = 384;
    float* ws  = (float*)d_ws;
    float* ln1 = ws;                        // [R,384]
    float* xzb = ln1 + (size_t)R * C;       // [R,768]
    float* xcb = xzb + (size_t)R * 768;     // [R,384]
    float* dbc = xcb + (size_t)R * C;       // [R,64] (56 used, padded)
    float* dtb = dbc + (size_t)R * 64;      // [R,384]
    float* ysb = dtb + (size_t)R * C;       // [R,384]
    float* yg  = ysb + (size_t)R * C;       // [R,384]
    float* x1  = yg  + (size_t)R * C;       // [R,384]
    float* h2  = x1  + (size_t)R * C;       // [R,384]
    float* m1  = h2  + (size_t)R * C;       // [R,384]

    dim3 blk(256);
    auto gg = [](int M, int N) { return dim3((unsigned)((N + BN - 1) / BN),
                                             (unsigned)((M + BM - 1) / BM)); };

    // 1) LN1
    ln_kernel<<<R, 128, 0, stream>>>(x, g1, b1, ln1, C);
    // 2) xz = ln1 @ W_in + b_in                       [4096,768]  (aligned)
    gemm_wmma_kernel<true><<<gg(R, 768), blk, 0, stream>>>(ln1, 384, W_in, 768, b_in,
                                                           nullptr, xzb, 768, R, 768, 384, 0);
    // 3) depthwise conv3d + SiLU on x-branch          [4096,384]
    conv_dw_silu_kernel<<<(R * C + 255) / 256, 256, 0, stream>>>(xzb, convw, convb, xcb);
    // 4) dbc = xc @ W_x                               [4096,56] (ldd=64, ragged N)
    gemm_wmma_kernel<false><<<gg(R, 56), blk, 0, stream>>>(xcb, 384, W_x, 56, nullptr,
                                                           nullptr, dbc, 64, R, 56, 384, 0);
    // 5) dt = softplus(dbc[:, :24] @ W_dt + b_dt)     [4096,384] (ragged K=24)
    gemm_wmma_kernel<false><<<gg(R, 384), blk, 0, stream>>>(dbc, 64, W_dt, 384, b_dt,
                                                            nullptr, dtb, 384, R, 384, 24, 1);
    // 6) selective scan over L=2048 per batch
    scan_kernel<<<dim3(24, 2), 256, 0, stream>>>(dtb, xcb, dbc, A_log, ysb);
    // 7) gate: LN(ys + xc*Dskip) * silu(z)
    gate_kernel<<<R, 128, 0, stream>>>(ysb, xcb, Dsk, xzb, on_g, on_b, yg);
    // 8) x1 = x + yg @ W_out + b_out                  (aligned)
    gemm_wmma_kernel<true><<<gg(R, 384), blk, 0, stream>>>(yg, 384, W_out, 384, b_out,
                                                           x, x1, 384, R, 384, 384, 0);
    // 9) LN2
    ln_kernel<<<R, 128, 0, stream>>>(x1, g2, b2, h2, C);
    // 10) m1 = gelu(h2 @ W1 + b1m)                    (aligned)
    gemm_wmma_kernel<true><<<gg(R, 384), blk, 0, stream>>>(h2, 384, W1, 384, b1m,
                                                           nullptr, m1, 384, R, 384, 384, 2);
    // 11) out = x1 + m1 @ W2 + b2m                    (aligned)
    gemm_wmma_kernel<true><<<gg(R, 384), blk, 0, stream>>>(m1, 384, W2, 384, b2m,
                                                           x1, out, 384, R, 384, 384, 0);
}
